// minLSTM_14998025797863
// MI455X (gfx1250) — matrix operand order repaired and checked by reading, not compiled, not run on hardware
//
#include <hip/hip_runtime.h>
#include <cmath>

// Problem dims (fixed by reference): x[B=4, S=4096, D=1024], W[D, 3D]
#define DDIM  1024
#define SEQ   4096
#define BATCH 4
#define MROWS (BATCH * SEQ)   // 16384
#define NCOLS (3 * DDIM)      // 3072
#define KDIM  DDIM            // 1024
#define NKITER (KDIM / 32)    // 32

typedef __attribute__((ext_vector_type(16))) __bf16 v16bf;
typedef __attribute__((ext_vector_type(8)))  __bf16 bf16x8;
typedef __attribute__((ext_vector_type(4)))  __bf16 bf16x4;
typedef __attribute__((ext_vector_type(8)))  float  v8f;

union Frag16 { v16bf v; bf16x8 h[2]; };

__device__ __forceinline__ v16bf lds_frag(const __bf16* p0, const __bf16* p1) {
    Frag16 f;
    f.h[0] = *reinterpret_cast<const bf16x8*>(p0);  // ds_load_b128
    f.h[1] = *reinterpret_cast<const bf16x8*>(p1);  // ds_load_b128
    return f.v;
}

__device__ __forceinline__ float softplusf(float z) {
    return fmaxf(z, 0.0f) + log1pf(__expf(-fabsf(z)));
}

__device__ __forceinline__ float logaddexpf_(float a, float b) {
    float mx = fmaxf(a, b);
    float mn = fminf(a, b);
    if (!(mn > -__builtin_inff()) || (mx - mn) > 32.0f) return mx;
    return mx + log1pf(__expf(mn - mx));
}

// ---------------------------------------------------------------------------
// Kernel 1: fused gate GEMM + log-gate epilogue.
// Block = 128 threads (4 wave32); tile = 128 rows x 32 cols per gate group.
// Each wave owns TWO 16-row m-tiles -> 36 WMMAs per K-step per wave against
// 24 B-frag + 8 A-frag ds_load_b128 (B fragments amortized over 2 m-tiles).
// fp32 operands split into bf16 hi/lo; 3x v_wmma_f32_16x16x32_bf16 per tile
// (hi*hi + hi*lo + lo*hi) ~= fp32 accuracy at bf16-pipe speed.
// Software pipeline: global loads for tile i+1 issued before WMMAs of tile i;
// LDS double-buffered (56 KB); packed bf16x4 (b64) LDS stores.
// ---------------------------------------------------------------------------
__global__ __launch_bounds__(128)
void gemm_gates_kernel(const float* __restrict__ x, const float* __restrict__ W,
                       float* __restrict__ logf_out, float* __restrict__ logv_out) {
    __shared__ __bf16 sAhi[2][128 * 32];
    __shared__ __bf16 sAlo[2][128 * 32];
    __shared__ __bf16 sBhi[2][3][32 * 32];   // transposed [n][k]
    __shared__ __bf16 sBlo[2][3][32 * 32];

    const int tid  = threadIdx.x;
    const int lane = tid & 31;
    const int wave = tid >> 5;               // 0..3, each owns 32 rows
    const int m0 = blockIdx.y * 128;
    const int n0 = blockIdx.x * 32;

    v8f acc[3][2][2] = {};   // [gate group][m-tile][n-tile]

    const int hsel = (lane < 16) ? 0 : 1;

    // Per-thread staging register mapping (128 threads):
    //  A: 8 float4 slots; slot = tid + i*128 -> row = slot/8, k = (slot%8)*4
    //  B: per group two (n, 4k) strips; n = tid&31, k = (tid>>5)*4 and +16
    int aRow[8];
    #pragma unroll
    for (int i = 0; i < 8; ++i) aRow[i] = (tid + i * 128) >> 3;
    const int aK  = (tid & 7) << 2;
    const int bN  = tid & 31;
    const int bK  = (tid >> 5) << 2;         // 0,4,8,12

    float4 aReg[8];
    float  bReg[3][2][4];

    auto load_tiles = [&](int k0) {
        #pragma unroll
        for (int i = 0; i < 8; ++i)
            aReg[i] = *reinterpret_cast<const float4*>(&x[(m0 + aRow[i]) * KDIM + k0 + aK]);
        #pragma unroll
        for (int g = 0; g < 3; ++g) {
            const float* wp = &W[(k0 + bK) * NCOLS + n0 + g * DDIM + bN];
            #pragma unroll
            for (int j = 0; j < 4; ++j) {
                bReg[g][0][j] = wp[j * NCOLS];
                bReg[g][1][j] = wp[(j + 16) * NCOLS];
            }
        }
    };

    auto store_tiles = [&](int buf) {
        #pragma unroll
        for (int i = 0; i < 8; ++i) {
            float vv[4] = { aReg[i].x, aReg[i].y, aReg[i].z, aReg[i].w };
            bf16x4 h4, l4;
            #pragma unroll
            for (int j = 0; j < 4; ++j) {
                __bf16 hi = (__bf16)vv[j];
                h4[j] = hi;
                l4[j] = (__bf16)(vv[j] - (float)hi);
            }
            *reinterpret_cast<bf16x4*>(&sAhi[buf][aRow[i] * 32 + aK]) = h4;  // ds_store_b64
            *reinterpret_cast<bf16x4*>(&sAlo[buf][aRow[i] * 32 + aK]) = l4;
        }
        #pragma unroll
        for (int g = 0; g < 3; ++g) {
            #pragma unroll
            for (int s = 0; s < 2; ++s) {
                bf16x4 h4, l4;
                #pragma unroll
                for (int j = 0; j < 4; ++j) {
                    __bf16 hi = (__bf16)bReg[g][s][j];
                    h4[j] = hi;
                    l4[j] = (__bf16)(bReg[g][s][j] - (float)hi);
                }
                *reinterpret_cast<bf16x4*>(&sBhi[buf][g][bN * 32 + bK + s * 16]) = h4;
                *reinterpret_cast<bf16x4*>(&sBlo[buf][g][bN * 32 + bK + s * 16]) = l4;
            }
        }
    };

    load_tiles(0);
    store_tiles(0);
    __syncthreads();

    for (int it = 0; it < NKITER; ++it) {
        const int cur = it & 1;

        if (it + 1 < NKITER)
            load_tiles((it + 1) * 32);      // global loads overlap WMMAs below

        // A fragments for this wave's two m-tiles
        v16bf a_hi[2], a_lo[2];
        #pragma unroll
        for (int mt = 0; mt < 2; ++mt) {
            int mrow = wave * 32 + mt * 16 + (lane & 15);
            const __bf16* aH = &sAhi[cur][mrow * 32 + hsel * 8];
            const __bf16* aL = &sAlo[cur][mrow * 32 + hsel * 8];
            a_hi[mt] = lds_frag(aH, aH + 16);
            a_lo[mt] = lds_frag(aL, aL + 16);
        }

        #pragma unroll
        for (int g = 0; g < 3; ++g) {
            #pragma unroll
            for (int t = 0; t < 2; ++t) {
                int nrow = t * 16 + (lane & 15);
                const __bf16* bH = &sBhi[cur][g][nrow * 32 + hsel * 8];
                const __bf16* bL = &sBlo[cur][g][nrow * 32 + hsel * 8];
                v16bf b_hi = lds_frag(bH, bH + 16);
                v16bf b_lo = lds_frag(bL, bL + 16);
                #pragma unroll
                for (int mt = 0; mt < 2; ++mt) {
                    acc[g][mt][t] = __builtin_amdgcn_wmma_f32_16x16x32_bf16(
                        false, a_hi[mt], false, b_hi, (short)0, acc[g][mt][t], false, false);
                    acc[g][mt][t] = __builtin_amdgcn_wmma_f32_16x16x32_bf16(
                        false, a_hi[mt], false, b_lo, (short)0, acc[g][mt][t], false, false);
                    acc[g][mt][t] = __builtin_amdgcn_wmma_f32_16x16x32_bf16(
                        false, a_lo[mt], false, b_hi, (short)0, acc[g][mt][t], false, false);
                }
            }
        }

        if (it + 1 < NKITER)
            store_tiles(1 - cur);           // fill the other buffer

        __syncthreads();
    }

    // ---- epilogue: log-space normalized gates ----
    // C/D layout: lane<16 -> N=lane, M=r ; lane>=16 -> N=lane-16, M=8+r
    #pragma unroll
    for (int mt = 0; mt < 2; ++mt) {
        #pragma unroll
        for (int t = 0; t < 2; ++t) {
            #pragma unroll
            for (int r = 0; r < 8; ++r) {
                int m    = m0 + wave * 32 + mt * 16 + r + hsel * 8;
                int dcol = n0 + t * 16 + (lane & 15);
                float f  = acc[0][mt][t][r];
                float ig = acc[1][mt][t][r];
                float h  = acc[2][mt][t][r];
                float diff   = softplusf(-f) - softplusf(-ig);
                float log_f  = -softplusf(diff);
                float log_i  = -softplusf(-diff);
                float log_th = (h >= 0.0f) ? __logf(h + 0.5f) : -softplusf(-h);
                logf_out[m * DDIM + dcol] = log_f;
                logv_out[m * DDIM + dcol] = log_i + log_th;
            }
        }
    }
}

// ---------------------------------------------------------------------------
// Kernel 2: chunked log-space scan along S (memory-bound).
// ---------------------------------------------------------------------------
__global__ __launch_bounds__(256)
void scan_kernel(const float* __restrict__ lf_in, const float* lv_in, float* h_out) {
    const int CHUNKS = 16, CLEN = 256, CHPB = 16;
    __shared__ float sA[CHPB][CHUNKS + 1];
    __shared__ float sV[CHPB][CHUNKS + 1];
    __shared__ float sC[CHPB][CHUNKS + 1];

    int tid   = threadIdx.x;
    int chl   = tid & 15;
    int chunk = tid >> 4;
    int c = blockIdx.x * CHPB + chl;
    int b = c >> 10;
    int d = c & (DDIM - 1);
    int base = (b * SEQ + chunk * CLEN) * DDIM + d;

    float A = 0.0f;
    float V = -__builtin_inff();
    for (int s = 0; s < CLEN; ++s) {
        float lf = lf_in[base + s * DDIM];
        float lv = lv_in[base + s * DDIM];
        A += lf;
        V = logaddexpf_(lf + V, lv);
    }
    sA[chl][chunk] = A;
    sV[chl][chunk] = V;
    __syncthreads();

    if (tid < CHPB) {
        float carry = -__builtin_inff();
        #pragma unroll
        for (int cc = 0; cc < CHUNKS; ++cc) {
            sC[tid][cc] = carry;
            carry = logaddexpf_(sA[tid][cc] + carry, sV[tid][cc]);
        }
    }
    __syncthreads();

    float lh = sC[chl][chunk];
    for (int s = 0; s < CLEN; ++s) {
        float lf = lf_in[base + s * DDIM];
        float lv = lv_in[base + s * DDIM];
        lh = logaddexpf_(lf + lh, lv);
        h_out[base + s * DDIM] = __expf(lh);
    }
}

extern "C" void kernel_launch(void* const* d_in, const int* in_sizes, int n_in,
                              void* d_out, int out_size, void* d_ws, size_t ws_size,
                              hipStream_t stream) {
    (void)in_sizes; (void)n_in; (void)out_size; (void)ws_size;
    const float* x = (const float*)d_in[0];
    const float* W = (const float*)d_in[1];
    float* out      = (float*)d_out;
    float* logf_buf = (float*)d_ws;   // 16384*1024 floats = 64 MB scratch
    float* logv_buf = out;            // reuse output buffer; scan overwrites it

    dim3 grid1(DDIM / 32, MROWS / 128);   // (32, 128)
    gemm_gates_kernel<<<grid1, 128, 0, stream>>>(x, W, logf_buf, logv_buf);

    scan_kernel<<<(BATCH * DDIM) / 16, 256, 0, stream>>>(logf_buf, logv_buf, out);
}
